// Flashback_18683107737866
// MI455X (gfx1250) — matrix operand order repaired and checked by reading, not compiled, not run on hardware
//
#include <hip/hip_runtime.h>
#include <hip/hip_bf16.h>

// ---------------------------------------------------------------------------
// Problem constants (from the reference)
// ---------------------------------------------------------------------------
#define N_USERS   10000
#define N_ITEMS   30000
#define HID       256
#define N_EDGES   500000
#define ATTN_HID  32
#define THRESHOLD 0.8f
#define NN        (N_USERS + N_ITEMS)       // 40000 combined nodes
#define TILES_E   (N_EDGES / 16)            // 31250 (divides exactly)

// Workspace layout (bytes, all 256-aligned). Total ~60.9 MB.
#define OFF_EWRAW 0ul                       // E floats: raw sigmoid weights
#define OFF_DEG   2000128ul                 // NN floats
#define OFF_CNT   2160128ul                 // uint: #edges above threshold
#define OFF_FB    2160132ul                 // int : fallback argmax edge
#define OFF_XW    2160384ul                 // NN*HID floats: x @ gcn_W
#define OFF_HX    43120384ul                // NN*HID halves: f16 concat(user,item)
#define OFF_HW1T  63600384ul                // 32*512 halves: W1^T f16
#define OFF_HWT   63633152ul                // 256*256 halves: gcn_W^T f16
// end: 63764224

typedef __attribute__((ext_vector_type(16))) _Float16 v16h;
typedef __attribute__((ext_vector_type(8)))  _Float16 v8h;
typedef __attribute__((ext_vector_type(4)))  _Float16 v4h;
typedef __attribute__((ext_vector_type(8)))  float    v8f;

// ---------------------------------------------------------------------------
// 0a) init: deg = 1.0 (self loop), counters = 0
// ---------------------------------------------------------------------------
__global__ void fb_init_kernel(float* __restrict__ deg, unsigned int* __restrict__ cnt,
                               int* __restrict__ fb) {
  int i = blockIdx.x * blockDim.x + threadIdx.x;
  if (i < NN) deg[i] = 1.0f;
  if (i == 0) { *cnt = 0u; *fb = -1; }
}

// ---------------------------------------------------------------------------
// 0b) pre-convert embeddings to f16 once: hx = concat(user,item) as half.
//     One float4 -> v4h per thread; boundary (N_USERS*HID) is 4-aligned.
// ---------------------------------------------------------------------------
__global__ void cvt_x_kernel(const float* __restrict__ u, const float* __restrict__ it,
                             _Float16* __restrict__ hx) {
  int i = blockIdx.x * blockDim.x + threadIdx.x;     // over NN*HID/4
  if (i >= NN * HID / 4) return;
  size_t off = (size_t)i * 4;
  const float* src = (off < (size_t)N_USERS * HID)
      ? (u + off) : (it + (off - (size_t)N_USERS * HID));
  float4 f = *(const float4*)src;
  v4h h;
  h[0] = (_Float16)f.x; h[1] = (_Float16)f.y;
  h[2] = (_Float16)f.z; h[3] = (_Float16)f.w;
  *(v4h*)(hx + off) = h;
}

// 0c) pre-transpose + convert W1 -> f16 [n][k] (32 x 512)
__global__ void cvt_w1_kernel(const float* __restrict__ W1, _Float16* __restrict__ hW1t) {
  int i = blockIdx.x * blockDim.x + threadIdx.x;     // over 512*32
  if (i >= 512 * ATTN_HID) return;
  int k = i >> 5, n = i & 31;                        // coalesced read W1[k][n]
  hW1t[n * 512 + k] = (_Float16)W1[i];
}

// 0d) pre-transpose + convert gcn_W -> f16 [n][k] (256 x 256)
__global__ void cvt_gw_kernel(const float* __restrict__ W, _Float16* __restrict__ hWt) {
  int i = blockIdx.x * blockDim.x + threadIdx.x;     // over 256*256
  if (i >= HID * HID) return;
  int k = i >> 8, n = i & 255;                       // coalesced read W[k][n]
  hWt[n * HID + k] = (_Float16)W[i];
}

// ---------------------------------------------------------------------------
// 1) Attention MLP via WMMA. One wave = 16 edges.
//    A: 16-byte f16 chunks gathered from hx (no conversion in loop).
//    B: W1^T f16 staged in LDS by plain uint4 copies.
// ---------------------------------------------------------------------------
__global__ __launch_bounds__(256)
void attn_kernel(const _Float16* __restrict__ hx, const int* __restrict__ eidx,
                 const _Float16* __restrict__ hW1t, const float* __restrict__ b1,
                 const float* __restrict__ W2, const float* __restrict__ b2,
                 float* __restrict__ ew_raw, unsigned int* __restrict__ cnt) {
  __shared__ __attribute__((aligned(32))) _Float16 W1t[ATTN_HID * 512];   // 32 KB
  const int t = threadIdx.x;
  for (int i = t; i < (ATTN_HID * 512) / 8; i += 256)      // straight 16B copies
    ((uint4*)W1t)[i] = ((const uint4*)hW1t)[i];
  __syncthreads();

  const int lane = t & 31;
  const int tile = blockIdx.x * 8 + (t >> 5);
  if (tile >= TILES_E) return;               // uniform per wave

  const int m    = lane & 15;                // A row / C column within tile
  const int half = lane >> 4;
  const int e    = tile * 16 + m;
  const int s    = eidx[e];
  const int d    = eidx[N_EDGES + e];
  const _Float16* __restrict__ urow = hx + (size_t)s * HID;
  const _Float16* __restrict__ irow = hx + (size_t)(N_USERS + d) * HID;

  v8f c0 = {}; v8f c1 = {};
  #pragma unroll
  for (int s32 = 0; s32 < 16; ++s32) {       // K = 512 = 16 * 32
    const int kk  = s32 * 32;
    const int kg0 = kk + 16 * half;          // A: two contiguous 8-half chunks
    const int kg1 = kg0 + 8;
    v8h a0 = *(const v8h*)((kg0 < HID) ? (urow + kg0) : (irow + (kg0 - HID)));
    v8h a1 = *(const v8h*)((kg1 < HID) ? (urow + kg1) : (irow + (kg1 - HID)));
    v16h a = __builtin_shufflevector(a0, a1, 0, 1, 2, 3, 4, 5, 6, 7,
                                             8, 9, 10, 11, 12, 13, 14, 15);
    const int kb = kk + 16 * half;           // B: 16 consecutive K per lane
    v16h bv0 = *(const v16h*)(&W1t[(size_t)(m     ) * 512 + kb]);
    v16h bv1 = *(const v16h*)(&W1t[(size_t)(m + 16) * 512 + kb]);
    c0 = __builtin_amdgcn_wmma_f32_16x16x32_f16(false, a, false, bv0, (short)0, c0, false, false);
    c1 = __builtin_amdgcn_wmma_f32_16x16x32_f16(false, a, false, bv1, (short)0, c1, false, false);
  }

  // Epilogue: lane holds hdn[M = r + 8*half][N = m] in c0 and N = m+16 in c1.
  const float bb0 = b1[m], bb1 = b1[m + 16];
  const float w20 = W2[m], w21 = W2[m + 16];
  const float bias2 = b2[0];
  float part[8];
  #pragma unroll
  for (int r = 0; r < 8; ++r) {
    float h0 = c0[r] + bb0; h0 = h0 > 0.0f ? h0 : 0.2f * h0;
    float h1 = c1[r] + bb1; h1 = h1 > 0.0f ? h1 : 0.2f * h1;
    part[r] = h0 * w20 + h1 * w21;
  }
  #pragma unroll
  for (int off = 1; off < 16; off <<= 1) {   // reduce within each 16-lane half
    #pragma unroll
    for (int r = 0; r < 8; ++r) part[r] += __shfl_xor(part[r], off, 32);
  }
  if (m == 0) {                              // lanes 0 and 16 write 8 rows each
    unsigned int nkeep = 0;
    #pragma unroll
    for (int r = 0; r < 8; ++r) {
      float z = part[r] + bias2;
      float w = 1.0f / (1.0f + expf(-z));
      ew_raw[tile * 16 + half * 8 + r] = w;
      if (w > THRESHOLD) ++nkeep;
    }
    if (nkeep) atomicAdd(cnt, nkeep);
  }
}

// ---------------------------------------------------------------------------
// 2) Fallback argmax (only matters if nothing passed the threshold).
//    Deterministic first-max tie-break like jnp.argmax.
// ---------------------------------------------------------------------------
__global__ __launch_bounds__(1024)
void fallback_kernel(const float* __restrict__ ew_raw, const unsigned int* __restrict__ cnt,
                     int* __restrict__ fb) {
  __shared__ float sw[1024];
  __shared__ int   si[1024];
  const int t = threadIdx.x;
  if (*cnt != 0u) { if (t == 0) *fb = -1; return; }
  float best = -1.0f; int bidx = 0;
  for (int i = t; i < N_EDGES; i += 1024) {
    float w = ew_raw[i];
    if (w > best) { best = w; bidx = i; }    // per-thread indices ascending
  }
  sw[t] = best; si[t] = bidx;
  __syncthreads();
  for (int s = 512; s > 0; s >>= 1) {
    if (t < s) {
      if (sw[t + s] > sw[t] || (sw[t + s] == sw[t] && si[t + s] < si[t])) {
        sw[t] = sw[t + s]; si[t] = si[t + s];
      }
    }
    __syncthreads();
  }
  if (t == 0) *fb = si[0];
}

// ---------------------------------------------------------------------------
// 3) Finalize edge weights + accumulate in-degree (exact: adds of 1.0f).
// ---------------------------------------------------------------------------
__global__ void finalize_kernel(const float* __restrict__ ew_raw, const int* __restrict__ eidx,
                                const int* __restrict__ fb, float* __restrict__ out_ew,
                                float* __restrict__ deg) {
  int e = blockIdx.x * blockDim.x + threadIdx.x;
  if (e >= N_EDGES) return;
  float w = ew_raw[e];
  bool keep = (w > THRESHOLD) || (e == *fb);
  out_ew[e] = keep ? w : 0.0f;
  if (keep) atomicAdd(&deg[eidx[N_EDGES + e]], 1.0f);
}

// ---------------------------------------------------------------------------
// 4) xw = hx @ gcn_W via WMMA.
//    Block: 64 rows x 128 cols (one N-half), 8 waves, each wave 16x64.
//    gcn_W^T f16 half staged in 64 KB LDS by plain uint4 copies.
// ---------------------------------------------------------------------------
__global__ __launch_bounds__(256)
void xw_kernel(const _Float16* __restrict__ hx, const _Float16* __restrict__ hWt,
               float* __restrict__ xw) {
  __shared__ __attribute__((aligned(32))) _Float16 Wt[128 * HID];   // 64 KB
  const int t = threadIdx.x;
  const int mgroup = blockIdx.x;             // 625 groups of 64 rows
  const int nhalf  = blockIdx.y;             // 2 halves of 128 cols
  const uint4* __restrict__ wsrc = (const uint4*)(hWt + (size_t)nhalf * 128 * HID);
  for (int i = t; i < (128 * HID) / 8; i += 256)           // straight 16B copies
    ((uint4*)Wt)[i] = wsrc[i];
  __syncthreads();

  const int lane = t & 31, wave = t >> 5;
  const int mt = wave >> 1;                  // 0..3 : 16-row tile in group
  const int nq = wave & 1;                   // 0..1 : 64-col quarter in half
  const int half = lane >> 4;
  const int m = mgroup * 64 + mt * 16 + (lane & 15);
  const _Float16* __restrict__ xrow = hx + (size_t)m * HID;

  v8f c[4] = {{}, {}, {}, {}};
  #pragma unroll
  for (int s32 = 0; s32 < 8; ++s32) {        // K = 256 = 8 * 32
    const int kk = s32 * 32;
    const int kg = kk + 16 * half;
    v8h a0 = *(const v8h*)(xrow + kg);
    v8h a1 = *(const v8h*)(xrow + kg + 8);
    v16h a = __builtin_shufflevector(a0, a1, 0, 1, 2, 3, 4, 5, 6, 7,
                                             8, 9, 10, 11, 12, 13, 14, 15);
    const int kb = kk + 16 * half;
    #pragma unroll
    for (int j = 0; j < 4; ++j) {            // 4 x (16x16) N tiles, A reused
      const int nl = nq * 64 + j * 16 + (lane & 15);
      v16h bv = *(const v16h*)(&Wt[(size_t)nl * HID + kb]);
      c[j] = __builtin_amdgcn_wmma_f32_16x16x32_f16(false, a, false, bv, (short)0, c[j], false, false);
    }
  }

  const int mbase = mgroup * 64 + mt * 16 + 8 * half;
  #pragma unroll
  for (int j = 0; j < 4; ++j) {
    const int n = nhalf * 128 + nq * 64 + j * 16 + (lane & 15);
    #pragma unroll
    for (int r = 0; r < 8; ++r)
      xw[(size_t)(mbase + r) * HID + n] = c[j][r];
  }
}

// ---------------------------------------------------------------------------
// 5) Self-loop term + bias: out = xw / deg + b
// ---------------------------------------------------------------------------
__global__ void outinit_kernel(const float* __restrict__ xw, const float* __restrict__ deg,
                               const float* __restrict__ gb, float* __restrict__ out) {
  int i = blockIdx.x * blockDim.x + threadIdx.x;
  if (i >= NN * HID) return;
  int node = i >> 8, h = i & 255;
  out[i] = xw[i] / deg[node] + gb[h];
}

// ---------------------------------------------------------------------------
// 6) Message scatter: out[dst] += dinv[src]*dinv[dst] * xw[src].
//    One wave per edge; 8 coalesced f32 atomics per lane (L2-resident).
// ---------------------------------------------------------------------------
__global__ __launch_bounds__(256)
void scatter_kernel(const float* __restrict__ ew_raw, const int* __restrict__ eidx,
                    const int* __restrict__ fb, const float* __restrict__ deg,
                    const float* __restrict__ xw, float* __restrict__ out) {
  const int lane = threadIdx.x & 31;
  const int e = blockIdx.x * 8 + (threadIdx.x >> 5);
  if (e >= N_EDGES) return;
  float w = ew_raw[e];
  if (!((w > THRESHOLD) || (e == *fb))) return;
  const int s = eidx[e];
  const int d = eidx[N_EDGES + e];
  const float norm = rsqrtf(deg[s]) * rsqrtf(deg[d]);
  const float* __restrict__ xs = xw + (size_t)s * HID;
  float* od = out + (size_t)d * HID;
  #pragma unroll
  for (int i = 0; i < 8; ++i) {
    const int h = lane + 32 * i;
    atomicAdd(&od[h], norm * xs[h]);
  }
}

// ---------------------------------------------------------------------------
extern "C" void kernel_launch(void* const* d_in, const int* in_sizes, int n_in,
                              void* d_out, int out_size, void* d_ws, size_t ws_size,
                              hipStream_t stream) {
  const float* uemb = (const float*)d_in[0];
  const float* iemb = (const float*)d_in[1];
  const int*   eidx = (const int*)d_in[2];
  const float* W1   = (const float*)d_in[3];
  const float* b1   = (const float*)d_in[4];
  const float* W2   = (const float*)d_in[5];
  const float* b2   = (const float*)d_in[6];
  const float* gW   = (const float*)d_in[7];
  const float* gb   = (const float*)d_in[8];

  float* out_gcn = (float*)d_out;                       // [NN*HID]
  float* out_ew  = out_gcn + (size_t)NN * HID;          // [N_EDGES]

  char* ws = (char*)d_ws;                               // needs ~61 MB
  float*        ew_raw = (float*)(ws + OFF_EWRAW);
  float*        deg    = (float*)(ws + OFF_DEG);
  unsigned int* cnt    = (unsigned int*)(ws + OFF_CNT);
  int*          fb     = (int*)(ws + OFF_FB);
  float*        xw     = (float*)(ws + OFF_XW);
  _Float16*     hx     = (_Float16*)(ws + OFF_HX);
  _Float16*     hW1t   = (_Float16*)(ws + OFF_HW1T);
  _Float16*     hWt    = (_Float16*)(ws + OFF_HWT);

  fb_init_kernel<<<(NN + 255) / 256, 256, 0, stream>>>(deg, cnt, fb);
  cvt_x_kernel<<<(NN * HID / 4 + 255) / 256, 256, 0, stream>>>(uemb, iemb, hx);
  cvt_w1_kernel<<<(512 * ATTN_HID + 255) / 256, 256, 0, stream>>>(W1, hW1t);
  cvt_gw_kernel<<<(HID * HID + 255) / 256, 256, 0, stream>>>(gW, hWt);

  attn_kernel<<<(TILES_E + 7) / 8, 256, 0, stream>>>(hx, eidx, hW1t, b1, W2, b2,
                                                     ew_raw, cnt);

  fallback_kernel<<<1, 1024, 0, stream>>>(ew_raw, cnt, fb);

  finalize_kernel<<<(N_EDGES + 255) / 256, 256, 0, stream>>>(ew_raw, eidx, fb, out_ew, deg);

  dim3 ggrid(625, 2);
  xw_kernel<<<ggrid, 256, 0, stream>>>(hx, hWt, xw);

  outinit_kernel<<<(NN * HID + 255) / 256, 256, 0, stream>>>(xw, deg, gb, out_gcn);

  scatter_kernel<<<(N_EDGES + 7) / 8, 256, 0, stream>>>(ew_raw, eidx, fb, deg, xw, out_gcn);
}
